// TELSTM_41120016892612
// MI455X (gfx1250) — matrix operand order repaired and checked by reading, not compiled, not run on hardware
//
#include <hip/hip_runtime.h>
#include <hip/hip_bf16.h>
#include <stdint.h>

typedef __bf16 bf16;
typedef __attribute__((ext_vector_type(16))) __bf16 v16bf;
typedef __attribute__((ext_vector_type(8)))  __bf16 v8bf;
typedef __attribute__((ext_vector_type(8)))  float  v8f;
typedef int v4i __attribute__((vector_size(16)));

#define B_DIM 1024
#define T_DIM 128
#define I_DIM 128
#define H_DIM 1024
#define O_DIM 64
#define KT    (H_DIM + I_DIM)   // 1152 combined K (hidden + input)
#define CH    32                 // K chunk per WMMA
#define NCH   (KT / CH)          // 36
#define BM    128                // block rows (batch)
#define BN    32                 // block cols (hidden units) -> 8 accums/wave

// ---- async copy global->LDS, guarded (falls back to ld+ds_store) ----------
#if defined(__HIP_DEVICE_COMPILE__) && \
    __has_builtin(__builtin_amdgcn_global_load_async_to_lds_b128) && \
    __has_builtin(__builtin_amdgcn_s_wait_asynccnt)
#define HAS_ASYNC_LDS 1
#else
#define HAS_ASYNC_LDS 0
#endif

#if HAS_ASYNC_LDS
typedef __attribute__((address_space(1))) v4i* gas_v4i_ptr;
typedef __attribute__((address_space(3))) v4i* las_v4i_ptr;
#endif

__device__ __forceinline__ void cp16(void* lds_dst, const void* gsrc) {
#if HAS_ASYNC_LDS
  __builtin_amdgcn_global_load_async_to_lds_b128(
      (gas_v4i_ptr)(uintptr_t)gsrc,
      (las_v4i_ptr)(uint32_t)(uintptr_t)lds_dst,
      0, 0);
#else
  *(int4*)lds_dst = *(const int4*)gsrc;
#endif
}

#if HAS_ASYNC_LDS
#define ASYNC_WAIT(n) __builtin_amdgcn_s_wait_asynccnt(n)
#else
#define ASYNC_WAIT(n) ((void)0)
#endif

__device__ __forceinline__ float sigm(float x) {
  return 1.0f / (1.0f + __expf(-x));
}
__device__ __forceinline__ float tanh_fast(float x) {
  x = fminf(fmaxf(x, -15.0f), 15.0f);
  float e = __expf(2.0f * x);
  return (e - 1.0f) / (e + 1.0f);
}

// ---------------------------------------------------------------------------
// One LSTM timestep: gates = [h_prev, x_t] @ Wc^T + (b_ih+b_hh); cell update.
// Grid: (B/BM=8, H/BN=32), 256 threads = 8 waves. Each wave owns 16 batch
// rows x 32 hidden cols x all 4 gates (8 accumulators = 64 VGPRs) so the
// pointwise update is register-local and the whole wave state fits <256 VGPR.
// ---------------------------------------------------------------------------
__global__ __launch_bounds__(256) void lstm_step_kernel(
    const bf16* __restrict__ xbf,    // [B][T*I] bf16
    const bf16* __restrict__ Wc,     // [4H][KT] bf16 (w_hh | w_ih per row)
    const float* __restrict__ b_ih, const float* __restrict__ b_hh,
    const bf16* __restrict__ hprev,  // [B][H]
    bf16* __restrict__ hnext,        // [B][H]
    float* __restrict__ c_state,     // [B][H]
    int t)
{
  __shared__ __align__(64) bf16 sA[2][BM][CH];       // 16 KB
  __shared__ __align__(64) bf16 sW[2][4][BN][CH];    // 16 KB

  const int tid = threadIdx.x;
  const int bm0 = blockIdx.x * BM;
  const int bn0 = blockIdx.y * BN;

  auto stage = [&](int buf, int kc) {
    const int k0 = kc * CH;
    {
      // A tile: 128 rows x 32 k, 32 bytes per thread (2 async b128)
      const int row = tid >> 1;
      const int off = (tid & 1) * 16;
      const bf16* asrc = (k0 < H_DIM)
          ? (hprev + (size_t)(bm0 + row) * H_DIM + k0)
          : (xbf + (size_t)(bm0 + row) * (T_DIM * I_DIM) + t * I_DIM + (k0 - H_DIM));
      cp16(&sA[buf][row][off],     asrc + off);
      cp16(&sA[buf][row][off + 8], asrc + off + 8);
    }
    {
      // W tile: 4 gates x 32 cols x 32 k = 128 rows, 32 bytes per thread
      const int wrow = tid >> 1;        // 0..127
      const int g    = wrow >> 5;       // 0..3
      const int col  = wrow & 31;       // 0..31
      const int off  = (tid & 1) * 16;
      const bf16* wsrc = Wc + (size_t)(g * H_DIM + bn0 + col) * KT + k0;
      cp16(&sW[buf][g][col][off],     wsrc + off);
      cp16(&sW[buf][g][col][off + 8], wsrc + off + 8);
    }
  };

  const int lane = tid & 31;
  const int wv   = tid >> 5;       // wave id 0..7, owns rows wv*16..+15
  const int mn   = lane & 15;
  const int half = lane >> 4;

  v8f acc[4][2] = {};              // [gate][col tile] 16x16 f32 accumulators

  stage(0, 0);

  for (int kc = 0; kc < NCH; ++kc) {
    const int buf = kc & 1;
    if (kc + 1 < NCH) {
      stage(buf ^ 1, kc + 1);
      ASYNC_WAIT(4);               // 4 async ops per wave per chunk
    } else {
      ASYNC_WAIT(0);
    }
    __syncthreads();

    // A fragment (16x32 bf16): per lane, contiguous k in [half*16, +16)
    const bf16* arow = &sA[buf][wv * 16 + mn][0];
    v8bf a0 = *(const v8bf*)(arow + half * 16);
    v8bf a1 = *(const v8bf*)(arow + half * 16 + 8);
    v16bf a = __builtin_shufflevector(a0, a1,
        0, 1, 2, 3, 4, 5, 6, 7, 8, 9, 10, 11, 12, 13, 14, 15);

#pragma unroll
    for (int g = 0; g < 4; ++g) {
#pragma unroll
      for (int ct = 0; ct < 2; ++ct) {
        // B fragment (32x16 bf16): per lane, contiguous k in [half*16, +16)
        v16bf b = *(const v16bf*)&sW[buf][g][ct * 16 + mn][half * 16];
        acc[g][ct] = __builtin_amdgcn_wmma_f32_16x16x32_bf16(
            false, a, false, b, (short)0, acc[g][ct], false, false);
      }
    }
    __syncthreads();
  }

  // LSTM pointwise update (gate order i, f, g, o as PyTorch)
#pragma unroll
  for (int ct = 0; ct < 2; ++ct) {
    const int hcol = bn0 + ct * 16 + mn;
    const float bi  = b_ih[0 * H_DIM + hcol] + b_hh[0 * H_DIM + hcol];
    const float bf_ = b_ih[1 * H_DIM + hcol] + b_hh[1 * H_DIM + hcol];
    const float bg  = b_ih[2 * H_DIM + hcol] + b_hh[2 * H_DIM + hcol];
    const float bo  = b_ih[3 * H_DIM + hcol] + b_hh[3 * H_DIM + hcol];
#pragma unroll
    for (int r = 0; r < 8; ++r) {
      const int brow = bm0 + wv * 16 + half * 8 + r;   // C/D layout: m = half*8+r
      const size_t idx = (size_t)brow * H_DIM + hcol;
      const float ig = sigm(acc[0][ct][r] + bi);
      const float fg = sigm(acc[1][ct][r] + bf_);
      const float gg = tanh_fast(acc[2][ct][r] + bg);
      const float og = sigm(acc[3][ct][r] + bo);
      const float cn = fg * c_state[idx] + ig * gg;
      const float hn = og * tanh_fast(cn);
      c_state[idx] = cn;
      hnext[idx] = (bf16)hn;
    }
  }
}

// ---------------------------------------------------------------------------
// Final FC: out = h_T @ w_fc^T + b_fc   (M=1024, N=64, K=1024)
// ---------------------------------------------------------------------------
__global__ __launch_bounds__(256) void fc_kernel(
    const bf16* __restrict__ h, const bf16* __restrict__ wfc,
    const float* __restrict__ bfc, float* __restrict__ out)
{
  __shared__ __align__(64) bf16 sA[2][BM][CH];        // 16 KB
  __shared__ __align__(64) bf16 sW[2][O_DIM][CH];     // 8 KB
  const int tid = threadIdx.x;
  const int bm0 = blockIdx.x * BM;

  auto stage = [&](int buf, int kc) {
    const int k0 = kc * CH;
    {
      const int row = tid >> 1;
      const int off = (tid & 1) * 16;
      const bf16* asrc = h + (size_t)(bm0 + row) * H_DIM + k0;
      cp16(&sA[buf][row][off],     asrc + off);
      cp16(&sA[buf][row][off + 8], asrc + off + 8);
    }
    {
      const int row = tid >> 2;        // 0..63
      const int off = (tid & 3) * 8;   // 16B per thread
      cp16(&sW[buf][row][off], wfc + (size_t)row * H_DIM + k0 + off);
    }
  };

  const int lane = tid & 31;
  const int wv   = tid >> 5;
  const int mn   = lane & 15;
  const int half = lane >> 4;

  v8f acc[4] = {};
  const int KC = H_DIM / CH;  // 32
  stage(0, 0);
  for (int kc = 0; kc < KC; ++kc) {
    const int buf = kc & 1;
    if (kc + 1 < KC) {
      stage(buf ^ 1, kc + 1);
      ASYNC_WAIT(3);
    } else {
      ASYNC_WAIT(0);
    }
    __syncthreads();

    const bf16* arow = &sA[buf][wv * 16 + mn][0];
    v8bf a0 = *(const v8bf*)(arow + half * 16);
    v8bf a1 = *(const v8bf*)(arow + half * 16 + 8);
    v16bf a = __builtin_shufflevector(a0, a1,
        0, 1, 2, 3, 4, 5, 6, 7, 8, 9, 10, 11, 12, 13, 14, 15);
#pragma unroll
    for (int ct = 0; ct < 4; ++ct) {
      v16bf b = *(const v16bf*)&sW[buf][ct * 16 + mn][half * 16];
      acc[ct] = __builtin_amdgcn_wmma_f32_16x16x32_bf16(
          false, a, false, b, (short)0, acc[ct], false, false);
    }
    __syncthreads();
  }

#pragma unroll
  for (int ct = 0; ct < 4; ++ct) {
    const int ocol = ct * 16 + mn;
    const float bb = bfc[ocol];
#pragma unroll
    for (int r = 0; r < 8; ++r) {
      const int brow = bm0 + wv * 16 + half * 8 + r;
      out[(size_t)brow * O_DIM + ocol] = acc[ct][r] + bb;
    }
  }
}

// ---------------------------------------------------------------------------
// Preparation kernels: fp32 -> bf16 conversion, weight packing, state init
// ---------------------------------------------------------------------------
__global__ void cvt_f32_bf16(const float* __restrict__ s, bf16* __restrict__ d, int n) {
  int i = blockIdx.x * blockDim.x + threadIdx.x;
  const int stride = gridDim.x * blockDim.x;
  for (; i < n; i += stride) d[i] = (bf16)s[i];
}

__global__ void pack_w(const float* __restrict__ w_hh, const float* __restrict__ w_ih,
                       bf16* __restrict__ Wc, int n) {
  int i = blockIdx.x * blockDim.x + threadIdx.x;
  const int stride = gridDim.x * blockDim.x;
  for (; i < n; i += stride) {
    const int g = i / KT, k = i % KT;
    const float v = (k < H_DIM) ? w_hh[g * H_DIM + k]
                                : w_ih[g * I_DIM + (k - H_DIM)];
    Wc[i] = (bf16)v;
  }
}

__global__ void init_state(bf16* __restrict__ h, float* __restrict__ c, int n) {
  int i = blockIdx.x * blockDim.x + threadIdx.x;
  const int stride = gridDim.x * blockDim.x;
  for (; i < n; i += stride) { h[i] = (bf16)0.0f; c[i] = 0.0f; }
}

// ---------------------------------------------------------------------------
extern "C" void kernel_launch(void* const* d_in, const int* in_sizes, int n_in,
                              void* d_out, int out_size, void* d_ws, size_t ws_size,
                              hipStream_t stream) {
  const float* x    = (const float*)d_in[0];
  const float* w_ih = (const float*)d_in[1];
  const float* w_hh = (const float*)d_in[2];
  const float* b_ih = (const float*)d_in[3];
  const float* b_hh = (const float*)d_in[4];
  const float* w_fc = (const float*)d_in[5];
  const float* b_fc = (const float*)d_in[6];
  float* out = (float*)d_out;

  char* p = (char*)d_ws;
  bf16*  xbf  = (bf16*)p;  p += (size_t)B_DIM * T_DIM * I_DIM * 2;  // 32 MB
  bf16*  Wc   = (bf16*)p;  p += (size_t)4 * H_DIM * KT * 2;         // 9.2 MB
  bf16*  wfcb = (bf16*)p;  p += (size_t)O_DIM * H_DIM * 2;          // 128 KB
  bf16*  h0   = (bf16*)p;  p += (size_t)B_DIM * H_DIM * 2;          // 2 MB
  bf16*  h1   = (bf16*)p;  p += (size_t)B_DIM * H_DIM * 2;          // 2 MB
  float* cst  = (float*)p; p += (size_t)B_DIM * H_DIM * 4;          // 4 MB

  cvt_f32_bf16<<<2048, 256, 0, stream>>>(x, xbf, B_DIM * T_DIM * I_DIM);
  cvt_f32_bf16<<<64,   256, 0, stream>>>(w_fc, wfcb, O_DIM * H_DIM);
  pack_w<<<2048, 256, 0, stream>>>(w_hh, w_ih, Wc, 4 * H_DIM * KT);
  init_state<<<512, 256, 0, stream>>>(h0, cst, B_DIM * H_DIM);

  for (int t = 0; t < T_DIM; ++t) {
    const bf16* hp = (t & 1) ? h1 : h0;
    bf16*       hn = (t & 1) ? h0 : h1;
    lstm_step_kernel<<<dim3(8, 32), 256, 0, stream>>>(
        xbf, Wc, b_ih, b_hh, hp, hn, cst, t);
  }
  // t=127 (odd) wrote into h0 -> final hidden state
  fc_kernel<<<8, 256, 0, stream>>>(h0, wfcb, b_fc, out);
}
